// GCNLayer_36962488549946
// MI455X (gfx1250) — matrix operand order repaired and checked by reading, not compiled, not run on hardware
//
#include <hip/hip_runtime.h>
#include <stdint.h>

#define NPTS 8192
#define BN_EPS 0.001f
#define NEG_SLOPE 0.2f

typedef __attribute__((ext_vector_type(16))) __bf16   bf16x16;
typedef __attribute__((ext_vector_type(8)))  float    f32x8;
typedef __attribute__((ext_vector_type(4)))  uint32_t u32x4;
typedef int v4i_ll __attribute__((vector_size(16)));   // matches builtin param

union BfVec { bf16x16 v; uint32_t u[8]; u32x4 q[2]; };

#if __has_builtin(__builtin_amdgcn_global_load_async_to_lds_b128) && \
    __has_builtin(__builtin_amdgcn_s_wait_asynccnt)
#define USE_ASYNC_LDS 1
#else
#define USE_ASYNC_LDS 0
#endif

#if USE_ASYNC_LDS
__device__ __forceinline__ __attribute__((address_space(3))) v4i_ll*
lds_cast(void* p) {
    // generic LDS pointer = aperture|offset; low 32 bits are the LDS offset
    return (__attribute__((address_space(3))) v4i_ll*)(uint32_t)(uintptr_t)p;
}
__device__ __forceinline__ __attribute__((address_space(1))) v4i_ll*
gas_cast(const void* p) {
    return (__attribute__((address_space(1))) v4i_ll*)(uintptr_t)p;
}
#endif

__device__ __forceinline__ uint32_t f2bf(float f) {
    uint32_t u = __float_as_uint(f);
    uint32_t r = u + 0x7FFFu + ((u >> 16) & 1u);   // round-to-nearest-even
    return r >> 16;
}

// ---------------------------------------------------------------------------
// Kernel 1: fp32 -> bf16 conversion + squared norms. One thread per point.
// ---------------------------------------------------------------------------
__global__ __launch_bounds__(256) void k_prep(const float* __restrict__ X,
                                              uint32_t* __restrict__ Xbf,
                                              float* __restrict__ sq) {
    const int p = blockIdx.x * 256 + threadIdx.x;   // 0..32767
    const float4* row = (const float4*)(X + (size_t)p * 64);
    float s = 0.f;
    u32x4 packed[8];
    #pragma unroll
    for (int c = 0; c < 16; ++c) {
        float4 f = row[c];
        s += f.x * f.x + f.y * f.y + f.z * f.z + f.w * f.w;
        uint32_t d0 = f2bf(f.x) | (f2bf(f.y) << 16);
        uint32_t d1 = f2bf(f.z) | (f2bf(f.w) << 16);
        packed[c >> 1][(c & 1) * 2 + 0] = d0;
        packed[c >> 1][(c & 1) * 2 + 1] = d1;
    }
    u32x4* drow = (u32x4*)(Xbf + (size_t)p * 32);
    #pragma unroll
    for (int c = 0; c < 8; ++c) drow[c] = packed[c];
    sq[p] = s;
}

// ---------------------------------------------------------------------------
// Kernel 2: fused X.X^T GEMM (bf16 WMMA, fp32 acc) + per-row top-8 selection.
// Grid: 4 batches * 64 row tiles = 256 WGs of 256 threads (8 waves).
// Each wave owns 16 rows; WG owns 128 rows; loop over 128 chunks of 64 cols.
// Column block staged into LDS with gfx1250 async direct-to-LDS copies.
// ---------------------------------------------------------------------------
__global__ __launch_bounds__(256) void k_knn(const uint32_t* __restrict__ Xbf,
                                             const float* __restrict__ sq,
                                             int* __restrict__ knn) {
    __shared__ __align__(16) uint32_t cb[64 * 36];     // col block, padded stride
    __shared__ float sqc[64];
    __shared__ __align__(16) float dist[128 * 68];     // inner products, padded

    const int b    = blockIdx.x >> 6;
    const int row0 = (blockIdx.x & 63) << 7;
    const int tid  = threadIdx.x;
    const int wave = tid >> 5;
    const int lane = tid & 31;
    const int l15  = lane & 15;
    const int lhi  = lane >> 4;

    // A tiles for this wave's 16 rows (K = 0..63), contiguous K order, same as B:
    // lane l (M = l&15) holds K in [16*lhi .. +15] per step.
    const int rowA = row0 + wave * 16 + l15;
    const u32x4* ab = (const u32x4*)(Xbf + ((size_t)(b * NPTS + rowA)) * 32);
    BfVec A0, A1;
    A0.q[0] = ab[2 * lhi + 0];  A0.q[1] = ab[2 * lhi + 1];     // K 0..31 region
    A1.q[0] = ab[2 * lhi + 4];  A1.q[1] = ab[2 * lhi + 5];     // K 32..63 region

    // per-thread staging addresses (fixed across chunks except global base)
    const int lin0 = tid * 4;            // dword index of 1st b128, j=0
    const int lin1 = 1024 + tid * 4;     // dword index of 2nd b128, j=1
    uint32_t* lds0 = cb + (lin0 >> 5) * 36 + (lin0 & 31);
    uint32_t* lds1 = cb + (lin1 >> 5) * 36 + (lin1 & 31);

    float tv[8]; int tg[8];
    #pragma unroll
    for (int q = 0; q < 8; ++q) { tv[q] = -1e30f; tg[q] = 0; }
    float tmin = -1e30f;

    for (int ch = 0; ch < 128; ++ch) {
        const int col0 = ch << 6;
        const uint32_t* gsrc = Xbf + ((size_t)(b * NPTS + col0)) * 32;
        __syncthreads();   // previous chunk's top-k done before restaging
        {
#if USE_ASYNC_LDS
            // gfx1250 async copy: global -> LDS directly, tracked by ASYNCcnt
            __builtin_amdgcn_global_load_async_to_lds_b128(
                gas_cast(gsrc + lin0), lds_cast(lds0), 0, 0);
            __builtin_amdgcn_global_load_async_to_lds_b128(
                gas_cast(gsrc + lin1), lds_cast(lds1), 0, 0);
#else
            *(u32x4*)lds0 = *(const u32x4*)(gsrc + lin0);
            *(u32x4*)lds1 = *(const u32x4*)(gsrc + lin1);
#endif
            if (tid < 64) sqc[tid] = sq[b * NPTS + col0 + tid];
            if (ch + 1 < 128) {  // warm L2->WGP path for next chunk
                __builtin_prefetch(gsrc + 2048 + lin0, 0, 0);
            }
        }
#if USE_ASYNC_LDS
        __builtin_amdgcn_s_wait_asynccnt(0);
#endif
        __syncthreads();

        // 4 subtiles of 16 cols: inner = A(16x64) * B(64x16), two K=32 WMMAs.
        #pragma unroll
        for (int ct = 0; ct < 4; ++ct) {
            const int col = ct * 16 + l15;             // B: lane n = l&15
            const uint32_t* bp = cb + col * 36 + lhi * 8;  // K = 16*lhi + e
            BfVec B0, B1;
            B0.q[0] = *(const u32x4*)(bp + 0);
            B0.q[1] = *(const u32x4*)(bp + 4);
            B1.q[0] = *(const u32x4*)(bp + 16);        // +32 bf16 -> K step 1
            B1.q[1] = *(const u32x4*)(bp + 20);
            f32x8 acc = {0.f, 0.f, 0.f, 0.f, 0.f, 0.f, 0.f, 0.f};
            acc = __builtin_amdgcn_wmma_f32_16x16x32_bf16(
                      false, A0.v, false, B0.v, (short)0, acc, false, false);
            acc = __builtin_amdgcn_wmma_f32_16x16x32_bf16(
                      false, A1.v, false, B1.v, (short)0, acc, false, false);
            // D layout: VGPR v -> M = v + 8*lhi, N = l&15
            const int rbase = wave * 16 + lhi * 8;
            #pragma unroll
            for (int v = 0; v < 8; ++v)
                dist[(rbase + v) * 68 + col] = acc[v];
        }
        __syncthreads();

        // Top-8 update: thread t owns row t. Rank by 2*inner - sq_col
        // (row's own sq term is constant per row -> cannot change ordering).
        if (tid < 128) {
            const float4* drow = (const float4*)(dist + tid * 68);
            #pragma unroll 4
            for (int c4 = 0; c4 < 16; ++c4) {
                float4 dv = drow[c4];
                #pragma unroll
                for (int e = 0; e < 4; ++e) {
                    int c = c4 * 4 + e;
                    float d = (e == 0) ? dv.x : (e == 1) ? dv.y
                             : (e == 2) ? dv.z : dv.w;
                    float score = 2.f * d - sqc[c];
                    if (score > tmin) {
                        int minpos = 0; float mv = tv[0];
                        #pragma unroll
                        for (int q = 1; q < 8; ++q)
                            if (tv[q] < mv) { mv = tv[q]; minpos = q; }
                        #pragma unroll
                        for (int q = 0; q < 8; ++q)
                            if (q == minpos) { tv[q] = score; tg[q] = col0 + c; }
                        float nm = tv[0];
                        #pragma unroll
                        for (int q = 1; q < 8; ++q) nm = fminf(nm, tv[q]);
                        tmin = nm;
                    }
                }
            }
        }
    }

    if (tid < 128) {
        int* orow = knn + ((size_t)(b * NPTS + row0 + tid)) * 8;
        #pragma unroll
        for (int q = 0; q < 8; ++q) orow[q] = tg[q];
    }
}

// ---------------------------------------------------------------------------
// Kernel 3: gather neighbors, fp32 edge GEMM (factored center term), BN,
// LeakyReLU, max over k. 4 points per WG; 64 threads (one per u) per point.
// ---------------------------------------------------------------------------
__global__ __launch_bounds__(256) void k_edge(const float* __restrict__ X,
                                              const float* __restrict__ W,
                                              const float* __restrict__ bn_g,
                                              const float* __restrict__ bn_b,
                                              const float* __restrict__ bn_m,
                                              const float* __restrict__ bn_v,
                                              const int* __restrict__ knn,
                                              float* __restrict__ out) {
    __shared__ float Wl[64 * 129];          // stride 129: conflict-free rows
    __shared__ float cent[4][64];
    __shared__ float diff[4][8][64];

    const int tid = threadIdx.x;
    for (int i = tid; i < 64 * 128; i += 256)
        Wl[(i >> 7) * 129 + (i & 127)] = W[i];

    const int sg = tid >> 6;
    const int u  = tid & 63;
    const int point = blockIdx.x * 4 + sg;
    const int b = point >> 13;
    const int n = point & 8191;

    {   // stage: thread role = channel c
        const int c = u;
        const size_t pbase = (size_t)(b * NPTS + n) * 64;
        float xi = X[pbase + c];
        cent[sg][c] = xi;
        const int* irow = knn + (size_t)(b * NPTS + n) * 8;
        #pragma unroll
        for (int k = 0; k < 8; ++k) {
            int j = irow[k];
            float xj = X[((size_t)(b * NPTS + j)) * 64 + c];
            diff[sg][k][c] = xj - xi;
        }
    }
    __syncthreads();

    const float inv = bn_g[u] * rsqrtf(bn_v[u] + BN_EPS);
    const float sh  = bn_b[u] - bn_m[u] * inv;
    const float* wr0 = Wl + u * 129;        // weights for (x_j - x_i)
    const float* wr1 = wr0 + 64;            // weights for center x_i

    float t2 = 0.f;
    #pragma unroll 8
    for (int c = 0; c < 64; ++c) t2 = fmaf(wr1[c], cent[sg][c], t2);

    float m = -1e30f;
    for (int k = 0; k < 8; ++k) {
        float t1 = 0.f;
        #pragma unroll 8
        for (int c = 0; c < 64; ++c) t1 = fmaf(wr0[c], diff[sg][k][c], t1);
        float val = fmaf(t1 + t2, inv, sh);
        val = (val >= 0.f) ? val : NEG_SLOPE * val;
        m = fmaxf(m, val);
    }
    out[((size_t)(b * NPTS + n)) * 64 + u] = m;
}

// ---------------------------------------------------------------------------
extern "C" void kernel_launch(void* const* d_in, const int* in_sizes, int n_in,
                              void* d_out, int out_size, void* d_ws, size_t ws_size,
                              hipStream_t stream) {
    (void)in_sizes; (void)n_in; (void)out_size; (void)ws_size;
    const float* X    = (const float*)d_in[0];
    const float* W    = (const float*)d_in[1];
    const float* bn_g = (const float*)d_in[2];
    const float* bn_b = (const float*)d_in[3];
    const float* bn_m = (const float*)d_in[4];
    const float* bn_v = (const float*)d_in[5];

    uint8_t* ws = (uint8_t*)d_ws;
    uint32_t* Xbf = (uint32_t*)ws;                               // 4 MiB (bf16 X)
    float*    sq  = (float*)(ws + 4u * 1024u * 1024u);           // 128 KiB
    int*      knn = (int*)(ws + 4u * 1024u * 1024u + 128u * 1024u); // 1 MiB

    k_prep<<<128,  256, 0, stream>>>(X, Xbf, sq);
    k_knn <<<256,  256, 0, stream>>>(Xbf, sq, knn);
    k_edge<<<8192, 256, 0, stream>>>(X, W, bn_g, bn_b, bn_m, bn_v, knn,
                                     (float*)d_out);
}
